// DefaultAttentionModule_85736137163522
// MI455X (gfx1250) — compile-verified, matrix operand order checked by reading, not run on hardware
//
#include <hip/hip_runtime.h>

#define NROWS 131072
#define DIN   1024
#define DHID  512
#define NBAGS 16
#define BN_EPS 1e-5f

typedef __attribute__((ext_vector_type(16))) __bf16        v16bf;
typedef __attribute__((ext_vector_type(8)))  float         v8f;
typedef __attribute__((ext_vector_type(8)))  unsigned int  v8u;

union Frag { v8u u; v16bf b; };

__device__ __forceinline__ unsigned short f2bf(float x) {
  unsigned int u = __float_as_uint(x);
  u += 0x7FFFu + ((u >> 16) & 1u);          // round-to-nearest-even
  return (unsigned short)(u >> 16);
}
__device__ __forceinline__ unsigned int pack2bf(float lo, float hi) {
  return (unsigned int)f2bf(lo) | ((unsigned int)f2bf(hi) << 16);
}

__device__ __forceinline__ void atomicMaxF(float* addr, float val) {
  unsigned int* ua = (unsigned int*)addr;
  unsigned int cur = *ua;
  while (__uint_as_float(cur) < val) {
    unsigned int prev = atomicCAS(ua, cur, __float_as_uint(val));
    if (prev == cur) break;
    cur = prev;
  }
}

__device__ __forceinline__ int find_bag(int row, const int* __restrict__ offs) {
  int bag = 0;
#pragma unroll
  for (int i = 1; i < NBAGS; ++i) bag += (row >= offs[i]) ? 1 : 0;
  return bag;
}

// ---------------------------------------------------------------- init
__global__ void k_init(float* colsum, float* colsumsq, float* bagmax,
                       float* bagsum, int* offs, const int* __restrict__ bag_sizes) {
  int t = threadIdx.x;                       // 512 threads
  if (t < DHID) { colsum[t] = 0.f; colsumsq[t] = 0.f; }
  if (t < 2 * NBAGS) { bagmax[t] = -__builtin_inff(); bagsum[t] = 0.f; }
  if (t == 0) {
    int a = 0; offs[0] = 0;
    for (int i = 0; i < NBAGS; ++i) { a += bag_sizes[i]; offs[i + 1] = a; }
  }
}

// ------------------------------------------------- W1 -> bf16 transpose
__global__ __launch_bounds__(256)
void k_w1t(const float* __restrict__ W1, unsigned short* __restrict__ W1t) {
  int idx = blockIdx.x * 256 + threadIdx.x;  // DIN*DHID threads
  int k = idx & (DIN - 1);
  int n = idx >> 10;
  W1t[(size_t)n * DIN + k] = f2bf(W1[(size_t)k * DHID + n]);
}

// ----------------------------------------------------- GEMM1 + BN stats
constexpr int BM = 128, BN = 64, BK = 32;
constexpr int ASTR = BK + 8;                 // LDS pad vs bank conflicts
constexpr int BSTR = BK + 8;

__global__ __launch_bounds__(256)
void k_gemm1(const float* __restrict__ A,            // features [NROWS][DIN]
             const unsigned short* __restrict__ Bt,  // W1^T bf16 [DHID][DIN]
             const float* __restrict__ bias,         // b1 [DHID]
             float* __restrict__ H,                  // out h [NROWS][DHID]
             float* __restrict__ colsum, float* __restrict__ colsumsq) {
  __shared__ unsigned short As[BM * ASTR];
  __shared__ unsigned short Bs[BN * BSTR];
  __shared__ float blkS[BN], blkS2[BN];

  const int tid  = threadIdx.x;
  const int wave = tid >> 5;
  const int lane = tid & 31;
  const int hf   = lane >> 4;                // K-half select
  const int l16  = lane & 15;
  const size_t row0 = (size_t)blockIdx.x * BM;
  const int n0 = blockIdx.y * BN;

  if (tid < BN) { blkS[tid] = 0.f; blkS2[tid] = 0.f; }

  v8f acc[4] = {};

  const int arow = tid >> 3;                 // 0..31
  const int akq  = (tid & 7) * 4;            // 0,4,..,28
  const int bnl  = tid >> 2;                 // 0..63
  const int bkq  = (tid & 3) * 8;            // 0,8,16,24

  // per-thread LDS byte address for the async B-tile copy (low 32 bits of the
  // generic pointer to a __shared__ object are the LDS-relative address)
  const unsigned ldsB = (unsigned)(size_t)&Bs[bnl * BSTR + bkq];
  const unsigned short* gB = Bt + (size_t)(n0 + bnl) * DIN + bkq;

  for (int k0 = 0; k0 < DIN; k0 += BK) {
    __syncthreads();
    // B tile: async global -> LDS copy (16B per lane), tracked on ASYNCcnt
    asm volatile("global_load_async_to_lds_b128 %0, %1, off"
                 :: "v"(ldsB), "v"(gB + k0) : "memory");
    // stage A tile (fp32 -> bf16), coalesced float4 loads
#pragma unroll
    for (int p = 0; p < 4; ++p) {
      int r = arow + p * 32;
      const float4 f = *(const float4*)(A + (row0 + r) * DIN + k0 + akq);
      *(unsigned int*)&As[r * ASTR + akq]     = pack2bf(f.x, f.y);
      *(unsigned int*)&As[r * ASTR + akq + 2] = pack2bf(f.z, f.w);
      if (k0 + BK < DIN)
        __builtin_prefetch(A + (row0 + r) * DIN + k0 + BK + akq, 0, 0);
    }
    asm volatile("s_wait_asynccnt 0x0" ::: "memory");
    __syncthreads();

    // A fragment: lane l16 = row M, halves split K per 16-bit WMMA layout
    Frag fa;
#pragma unroll
    for (int j = 0; j < 8; ++j) {
      int k = ((j & 3) << 1) + ((j >> 2) << 4) + hf * 8;
      fa.u[j] = *(const unsigned int*)&As[(wave * 16 + l16) * ASTR + k];
    }
#pragma unroll
    for (int t = 0; t < 4; ++t) {
      Frag fb;
#pragma unroll
      for (int j = 0; j < 8; ++j) {
        int k = ((j & 3) << 1) + ((j >> 2) << 4) + hf * 8;
        fb.u[j] = *(const unsigned int*)&Bs[(t * 16 + l16) * BSTR + k];
      }
      acc[t] = __builtin_amdgcn_wmma_f32_16x16x32_bf16(
          false, fa.b, false, fb.b, (short)0, acc[t], false, false);
    }
  }

  __syncthreads();

  // epilogue: +bias, store h, per-column sum / sumsq for BatchNorm
#pragma unroll
  for (int t = 0; t < 4; ++t) {
    int col = n0 + t * 16 + l16;
    float bb = bias[col];
    float s = 0.f, s2 = 0.f;
    size_t grow = row0 + wave * 16;
#pragma unroll
    for (int r = 0; r < 8; ++r) {
      int m = r + hf * 8;                    // C layout: lanes>=16 hold M+8
      float v = acc[t][r] + bb;
      H[(grow + m) * DHID + col] = v;
      s += v; s2 += v * v;
    }
    s  += __shfl_xor(s, 16, 32);
    s2 += __shfl_xor(s2, 16, 32);
    if (hf == 0) {
      atomicAdd(&blkS[t * 16 + l16], s);     // ds_add_f32
      atomicAdd(&blkS2[t * 16 + l16], s2);
    }
  }
  __syncthreads();
  if (tid < BN) {
    atomicAdd(&colsum[n0 + tid],   blkS[tid]);
    atomicAdd(&colsumsq[n0 + tid], blkS2[tid]);
  }
}

// ------------------------------------------------ fold BN params
__global__ void k_stats(const float* __restrict__ colsum, const float* __restrict__ colsumsq,
                        const float* __restrict__ gamma, const float* __restrict__ beta,
                        float* __restrict__ scale, float* __restrict__ shiftv) {
  int i = blockIdx.x * 256 + threadIdx.x;    // DHID threads
  float inv = 1.f / (float)NROWS;
  float mean = colsum[i] * inv;
  float var  = colsumsq[i] * inv - mean * mean;
  float sc   = rsqrtf(var + BN_EPS) * gamma[i];
  scale[i]  = sc;
  shiftv[i] = beta[i] - mean * sc;
}

// --------------------------------- BN apply + ReLU + GEMM2 + bag max
__global__ __launch_bounds__(256)
void k_rowscore(const float* __restrict__ H,
                const float* __restrict__ scale, const float* __restrict__ shiftv,
                const float* __restrict__ W2, const float* __restrict__ b2,
                const int* __restrict__ offs,
                float* __restrict__ scores, float* __restrict__ bagmax) {
  int wave = threadIdx.x >> 5, lane = threadIdx.x & 31;
  int row = blockIdx.x * 8 + wave;
  const float* hrow = H + (size_t)row * DHID;
  float a0 = 0.f, a1 = 0.f;
#pragma unroll
  for (int ch = 0; ch < 4; ++ch) {
    int c = ch * 128 + lane * 4;
    float4 v  = *(const float4*)(hrow + c);
    float4 sc = *(const float4*)(scale + c);
    float4 sh = *(const float4*)(shiftv + c);
    float4 wa = *(const float4*)(W2 + 2 * c);      // rows c, c+1
    float4 wb = *(const float4*)(W2 + 2 * c + 4);  // rows c+2, c+3
    float h0 = fmaxf(v.x * sc.x + sh.x, 0.f);
    float h1 = fmaxf(v.y * sc.y + sh.y, 0.f);
    float h2 = fmaxf(v.z * sc.z + sh.z, 0.f);
    float h3 = fmaxf(v.w * sc.w + sh.w, 0.f);
    a0 += h0 * wa.x + h1 * wa.z + h2 * wb.x + h3 * wb.z;
    a1 += h0 * wa.y + h1 * wa.w + h2 * wb.y + h3 * wb.w;
  }
#pragma unroll
  for (int m = 16; m >= 1; m >>= 1) {
    a0 += __shfl_xor(a0, m, 32);
    a1 += __shfl_xor(a1, m, 32);
  }
  if (lane == 0) {
    float s0 = a0 + b2[0], s1 = a1 + b2[1];
    scores[2 * row] = s0; scores[2 * row + 1] = s1;
    int bag = find_bag(row, offs);
    atomicMaxF(&bagmax[2 * bag],     s0);
    atomicMaxF(&bagmax[2 * bag + 1], s1);
  }
}

// ------------------------------------------ exp + per-bag sums
__global__ __launch_bounds__(256)
void k_expsum(const float* __restrict__ scores, const float* __restrict__ bagmax,
              const int* __restrict__ offs, float* __restrict__ ebuf,
              float* __restrict__ bagsum) {
  __shared__ float ls[2 * NBAGS];
  int t = threadIdx.x;
  if (t < 2 * NBAGS) ls[t] = 0.f;
  __syncthreads();
  int row = blockIdx.x * 256 + t;
  int bag = find_bag(row, offs);
  float e0 = expf(scores[2 * row]     - bagmax[2 * bag]);
  float e1 = expf(scores[2 * row + 1] - bagmax[2 * bag + 1]);
  ebuf[2 * row] = e0; ebuf[2 * row + 1] = e1;
  atomicAdd(&ls[2 * bag], e0);
  atomicAdd(&ls[2 * bag + 1], e1);
  __syncthreads();
  if (t < 2 * NBAGS) atomicAdd(&bagsum[t], ls[t]);
}

// ------------------------------------------ normalize
__global__ __launch_bounds__(256)
void k_final(const float* __restrict__ ebuf, const float* __restrict__ bagsum,
             const int* __restrict__ offs, float* __restrict__ out) {
  int row = blockIdx.x * 256 + threadIdx.x;
  int bag = find_bag(row, offs);
  out[2 * row]     = ebuf[2 * row]     / bagsum[2 * bag];
  out[2 * row + 1] = ebuf[2 * row + 1] / bagsum[2 * bag + 1];
}

extern "C" void kernel_launch(void* const* d_in, const int* in_sizes, int n_in,
                              void* d_out, int out_size, void* d_ws, size_t ws_size,
                              hipStream_t stream) {
  (void)in_sizes; (void)n_in; (void)out_size; (void)ws_size;
  const float* features = (const float*)d_in[0];
  const float* W1       = (const float*)d_in[1];
  const float* b1       = (const float*)d_in[2];
  const float* gamma    = (const float*)d_in[3];
  const float* beta     = (const float*)d_in[4];
  const float* W2       = (const float*)d_in[5];
  const float* b2       = (const float*)d_in[6];
  const int*   bag_sizes = (const int*)d_in[7];
  float* out = (float*)d_out;

  char* ws = (char*)d_ws;
  size_t off = 0;
  float* H = (float*)(ws + off);              off += (size_t)NROWS * DHID * sizeof(float);
  unsigned short* W1t = (unsigned short*)(ws + off); off += (size_t)DIN * DHID * 2;
  float* scores  = (float*)(ws + off);        off += (size_t)NROWS * 2 * sizeof(float);
  float* ebuf    = (float*)(ws + off);        off += (size_t)NROWS * 2 * sizeof(float);
  float* colsum  = (float*)(ws + off);        off += DHID * sizeof(float);
  float* colsumsq = (float*)(ws + off);       off += DHID * sizeof(float);
  float* scale   = (float*)(ws + off);        off += DHID * sizeof(float);
  float* shiftv  = (float*)(ws + off);        off += DHID * sizeof(float);
  float* bagmax  = (float*)(ws + off);        off += 2 * NBAGS * sizeof(float);
  float* bagsum  = (float*)(ws + off);        off += 2 * NBAGS * sizeof(float);
  int*   offs    = (int*)(ws + off);          off += (NBAGS + 1) * sizeof(int);

  k_init<<<1, 512, 0, stream>>>(colsum, colsumsq, bagmax, bagsum, offs, bag_sizes);
  k_w1t<<<(DIN * DHID) / 256, 256, 0, stream>>>(W1, W1t);
  k_gemm1<<<dim3(NROWS / BM, DHID / BN), 256, 0, stream>>>(features, W1t, b1, H,
                                                           colsum, colsumsq);
  k_stats<<<DHID / 256, 256, 0, stream>>>(colsum, colsumsq, gamma, beta, scale, shiftv);
  k_rowscore<<<NROWS / 8, 256, 0, stream>>>(H, scale, shiftv, W2, b2, offs, scores, bagmax);
  k_expsum<<<NROWS / 256, 256, 0, stream>>>(scores, bagmax, offs, ebuf, bagsum);
  k_final<<<NROWS / 256, 256, 0, stream>>>(ebuf, bagsum, offs, out);
}
